// BiPointNetCls_81484119540296
// MI455X (gfx1250) — 8-bit matrix operand order repaired and checked by reading, compiled, not run on hardware
//
#include <hip/hip_runtime.h>
#include <hip/hip_bf16.h>
#include <math.h>

typedef __attribute__((ext_vector_type(8))) int v8i;

#define BN_EPS 1e-5f

__global__ void __launch_bounds__(256) k_sign_pad(
    const float* __restrict__ x, signed char* __restrict__ y,
    int rows, int K, int Kpad) {
  long long i = (long long)blockIdx.x * blockDim.x + threadIdx.x;
  long long total = (long long)rows * Kpad;
  if (i >= total) return;
  int k = (int)(i % (long long)Kpad);
  long long r = i / Kpad;
  float v = (k < K) ? x[r * K + k] : 0.0f;
  y[i] = (signed char)((v > 0.0f) ? 1 : ((v < 0.0f) ? -1 : 0));
}

__device__ __forceinline__ v8i load_a_frag(const signed char* arow, int k0, int hi) {
  int4 a0 = *(const int4*)(arow + k0 + hi * 16);
  int4 a1 = *(const int4*)(arow + k0 + 32 + hi * 16);
  v8i r = {a0.x, a0.y, a0.z, a0.w, a1.x, a1.y, a1.z, a1.w};
  return r;
}

__device__ __forceinline__ v8i load_b_frag(const signed char* wrow, int k0, int hi) {
  int4 b0 = *(const int4*)(wrow + k0 + hi * 16);
  int4 b1 = *(const int4*)(wrow + k0 + 32 + hi * 16);
  v8i r = {b0.x, b0.y, b0.z, b0.w, b1.x, b1.y, b1.z, b1.w};
  return r;
}

__device__ __forceinline__ void store_tile(
    float* __restrict__ out, v8i acc, int tm, int hi, int col, int O,
    const float* __restrict__ bias) {
  if (col < O) {
    float bv = bias[col];
#pragma unroll
    for (int j = 0; j < 8; ++j) {
      int rowOut = tm * 16 + hi * 8 + j;
      out[(size_t)rowOut * O + col] = (float)acc[j] + bv;
    }
  }
}

__global__ void __launch_bounds__(256) k_bgemm_i8(
    const signed char* __restrict__ A, const signed char* __restrict__ W,
    const float* __restrict__ bias, float* __restrict__ out,
    int M, int Kpad, int O, int groupsO, int totalWaves) {
  int wave = blockIdx.x * 8 + (threadIdx.x >> 5);
  if (wave >= totalWaves) return;
  int lane = threadIdx.x & 31;
  int tm = wave / groupsO;
  int tg = wave % groupsO;
  int l15 = lane & 15;
  int hi  = lane >> 4;

  int rowA = tm * 16 + l15;
  const signed char* arow = A + (size_t)rowA * Kpad;

  int colBase = tg * 64;
  int c0 = colBase + l15,      c1 = colBase + 16 + l15;
  int c2 = colBase + 32 + l15, c3 = colBase + 48 + l15;
  int m0 = c0 < O ? c0 : O - 1, m1 = c1 < O ? c1 : O - 1;
  int m2 = c2 < O ? c2 : O - 1, m3 = c3 < O ? c3 : O - 1;
  const signed char* w0 = W + (size_t)m0 * Kpad;
  const signed char* w1 = W + (size_t)m1 * Kpad;
  const signed char* w2 = W + (size_t)m2 * Kpad;
  const signed char* w3 = W + (size_t)m3 * Kpad;

  v8i acc0 = {0, 0, 0, 0, 0, 0, 0, 0};
  v8i acc1 = acc0, acc2 = acc0, acc3 = acc0;

  for (int k0 = 0; k0 < Kpad; k0 += 64) {
    v8i am = load_a_frag(arow, k0, hi);
    v8i b0 = load_b_frag(w0, k0, hi);
    v8i b1 = load_b_frag(w1, k0, hi);
    v8i b2 = load_b_frag(w2, k0, hi);
    v8i b3 = load_b_frag(w3, k0, hi);
    if (k0 + 256 < Kpad)
      __builtin_prefetch((const void*)(arow + k0 + 256), 0, 3);
    acc0 = __builtin_amdgcn_wmma_i32_16x16x64_iu8(true, am, true, b0, acc0,
                                                  false, false);
    acc1 = __builtin_amdgcn_wmma_i32_16x16x64_iu8(true, am, true, b1, acc1,
                                                  false, false);
    acc2 = __builtin_amdgcn_wmma_i32_16x16x64_iu8(true, am, true, b2, acc2,
                                                  false, false);
    acc3 = __builtin_amdgcn_wmma_i32_16x16x64_iu8(true, am, true, b3, acc3,
                                                  false, false);
  }

  store_tile(out, acc0, tm, hi, c0, O, bias);
  store_tile(out, acc1, tm, hi, c1, O, bias);
  store_tile(out, acc2, tm, hi, c2, O, bias);
  store_tile(out, acc3, tm, hi, c3, O, bias);
}

__global__ void __launch_bounds__(256) k_bn_stats(
    const float* __restrict__ x, float* __restrict__ mean,
    float* __restrict__ inv, int Rw, int C) {
  int c = blockIdx.x;
  float s1 = 0.0f, s2 = 0.0f;
  for (int r = threadIdx.x; r < Rw; r += 256) {
    float v = x[(size_t)r * C + c];
    s1 += v;
    s2 += v * v;
  }
  __shared__ float a1[256];
  __shared__ float a2[256];
  a1[threadIdx.x] = s1;
  a2[threadIdx.x] = s2;
  __syncthreads();
  for (int s = 128; s > 0; s >>= 1) {
    if (threadIdx.x < s) {
      a1[threadIdx.x] += a1[threadIdx.x + s];
      a2[threadIdx.x] += a2[threadIdx.x + s];
    }
    __syncthreads();
  }
  if (threadIdx.x == 0) {
    float m = a1[0] / (float)Rw;
    float v = a2[0] / (float)Rw - m * m;
    mean[c] = m;
    inv[c] = rsqrtf(v + BN_EPS);
  }
}

__global__ void __launch_bounds__(256) k_bn_relu(
    float* __restrict__ x, const float* __restrict__ mean,
    const float* __restrict__ inv, const float* __restrict__ g,
    const float* __restrict__ b, long long total, int C) {
  long long i = (long long)blockIdx.x * 256 + threadIdx.x;
  if (i >= total) return;
  int c = (int)(i % (long long)C);
  float v = (x[i] - mean[c]) * inv[c] * g[c] + b[c];
  x[i] = v > 0.0f ? v : 0.0f;
}

__global__ void __launch_bounds__(256) k_bn_relu_sign(
    const float* __restrict__ xin, float* __restrict__ xout,
    signed char* __restrict__ sgn, const float* __restrict__ mean,
    const float* __restrict__ inv, const float* __restrict__ g,
    const float* __restrict__ b, long long rows, int C, int Kpad) {
  long long i = (long long)blockIdx.x * 256 + threadIdx.x;
  long long total = rows * Kpad;
  if (i >= total) return;
  int k = (int)(i % (long long)Kpad);
  long long r = i / Kpad;
  signed char s8 = 0;
  if (k < C) {
    float v = (xin[r * C + k] - mean[k]) * inv[k] * g[k] + b[k];
    v = v > 0.0f ? v : 0.0f;
    if (xout) xout[r * C + k] = v;
    s8 = (signed char)(v > 0.0f ? 1 : 0);
  }
  sgn[i] = s8;
}

__global__ void __launch_bounds__(256) k_maxpool(
    const float* __restrict__ x, float* __restrict__ out,
    int N, int C, float offset) {
  int c = blockIdx.x % C;
  int b = blockIdx.x / C;
  float m = -3.4e38f;
  for (int n = threadIdx.x; n < N; n += 256)
    m = fmaxf(m, x[((size_t)b * N + n) * C + c]);
  __shared__ float sm[256];
  sm[threadIdx.x] = m;
  __syncthreads();
  for (int s = 128; s > 0; s >>= 1) {
    if (threadIdx.x < s) sm[threadIdx.x] = fmaxf(sm[threadIdx.x], sm[threadIdx.x + s]);
    __syncthreads();
  }
  if (threadIdx.x == 0) out[(size_t)b * C + c] = sm[0] + offset;
}

__global__ void __launch_bounds__(256) k_bmm_eye(
    const float* __restrict__ h, const float* __restrict__ t,
    float* __restrict__ out, int N, int d, long long total) {
  long long i = (long long)blockIdx.x * 256 + threadIdx.x;
  if (i >= total) return;
  int j = (int)(i % (long long)d);
  long long rn = i / d;
  int b = (int)(rn / (long long)N);
  const float* hp = h + rn * d;
  const float* tp = t + (size_t)b * d * d;
  float s = 0.0f;
  for (int c = 0; c < d; ++c)
    s += hp[c] * (tp[c * d + j] + ((c == j) ? 1.0f : 0.0f));
  out[i] = s;
}

static inline int cdiv(long long a, long long b) { return (int)((a + b - 1) / b); }

extern "C" void kernel_launch(void* const* d_in, const int* in_sizes, int n_in,
                              void* d_out, int out_size, void* d_ws, size_t ws_size,
                              hipStream_t stream) {
  (void)in_sizes; (void)n_in; (void)out_size; (void)ws_size;
  const int B = 16, Npts = 4096;
  const int R = B * Npts;
  const float OFF = -3.5836f;

#define F(i) ((const float*)d_in[(i)])

  char* wp = (char*)d_ws;
  auto alloc = [&](size_t bytes) -> void* {
    void* q = (void*)wp;
    wp += (bytes + 255) & ~(size_t)255;
    return q;
  };
  float* bufA = (float*)alloc((size_t)R * 1024 * 4);
  float* bufB = (float*)alloc((size_t)R * 1024 * 4);
  float* bufC = (float*)alloc((size_t)R * 128 * 4);
  signed char* ai8 = (signed char*)alloc((size_t)R * 1024);
  signed char* wi8 = (signed char*)alloc((size_t)4096 * 1024);
  float* meanb = (float*)alloc(1024 * 4);
  float* invb  = (float*)alloc(1024 * 4);
  float* pool  = (float*)alloc((size_t)B * 1024 * 4);
  float* q1    = (float*)alloc((size_t)B * 512 * 4);
  float* q2    = (float*)alloc((size_t)B * 256 * 4);
  float* tmat  = (float*)alloc((size_t)B * 4096 * 4);

  auto sign_pad = [&](const float* src, signed char* dst, int rows, int K) -> int {
    int Kpad = (K + 63) & ~63;
    long long total = (long long)rows * Kpad;
    k_sign_pad<<<cdiv(total, 256), 256, 0, stream>>>(src, dst, rows, K, Kpad);
    return Kpad;
  };
  auto gemm = [&](int M, int Kpad, int O, const float* bias, float* outp) {
    int tilesM = (M + 15) / 16;
    int groupsO = (O + 63) / 64;
    int tw = tilesM * groupsO;
    k_bgemm_i8<<<(tw + 7) / 8, 256, 0, stream>>>(ai8, wi8, bias, outp,
                                                 M, Kpad, O, groupsO, tw);
  };
  auto biconv = [&](const float* in, int M, int K, const float* w,
                    const float* bias, int O, float* outp) {
    int Kpad = sign_pad(in, ai8, M, K);
    sign_pad(w, wi8, O, K);
    gemm(M, Kpad, O, bias, outp);
  };
  auto biconvP = [&](int M, int K, const float* w, const float* bias, int O,
                     float* outp) {
    int Kpad = (K + 63) & ~63;
    sign_pad(w, wi8, O, K);
    gemm(M, Kpad, O, bias, outp);
  };
  auto bnrelu = [&](float* h, int Rw, int C, const float* g, const float* b) {
    k_bn_stats<<<C, 256, 0, stream>>>(h, meanb, invb, Rw, C);
    long long total = (long long)Rw * C;
    k_bn_relu<<<cdiv(total, 256), 256, 0, stream>>>(h, meanb, invb, g, b, total, C);
  };
  auto bnrelu_sign = [&](const float* gsrc, float* fdst, long long Rw, int C,
                         const float* g, const float* b) {
    k_bn_stats<<<C, 256, 0, stream>>>(gsrc, meanb, invb, (int)Rw, C);
    int Kpad = (C + 63) & ~63;
    long long total = Rw * Kpad;
    k_bn_relu_sign<<<cdiv(total, 256), 256, 0, stream>>>(
        gsrc, fdst, ai8, meanb, invb, g, b, Rw, C, Kpad);
  };

  auto tnet = [&](bool preSigned, const float* hin, int d, const int* ix,
                  float* big, float* mid, float* tout) {
    if (preSigned) biconvP(R, d, F(ix[0]), F(ix[3]), 64, big);
    else           biconv(hin, R, d, F(ix[0]), F(ix[3]), 64, big);
    bnrelu_sign(big, nullptr, R, 64, F(ix[6]), F(ix[9]));
    biconvP(R, 64, F(ix[1]), F(ix[4]), 128, mid);
    bnrelu_sign(mid, nullptr, R, 128, F(ix[7]), F(ix[10]));
    biconvP(R, 128, F(ix[2]), F(ix[5]), 1024, big);
    bnrelu(big, R, 1024, F(ix[8]), F(ix[11]));
    k_maxpool<<<B * 1024, 256, 0, stream>>>(big, pool, Npts, 1024, OFF);
    biconv(pool, B, 1024, F(ix[12]), F(ix[14]), 512, q1);
    bnrelu_sign(q1, nullptr, B, 512, F(ix[16]), F(ix[18]));
    biconvP(B, 512, F(ix[13]), F(ix[15]), 256, q2);
    bnrelu_sign(q2, nullptr, B, 256, F(ix[17]), F(ix[19]));
    biconvP(B, 256, F(ix[20]), F(ix[21]), d * d, tout);
  };

  int ix1[22], ix2[22];
  for (int i = 0; i < 22; ++i) { ix1[i] = 1 + i; ix2[i] = 23 + i; }

  tnet(false, F(0), 3, ix1, bufA, bufC, tmat);
  {
    long long total = (long long)R * 3;
    k_bmm_eye<<<cdiv(total, 256), 256, 0, stream>>>(F(0), tmat, bufB, Npts, 3, total);
  }
  bnrelu_sign(bufB, nullptr, R, 3, F(45), F(46));

  biconvP(R, 3,  F(49), F(52), 64, bufA);
  bnrelu_sign(bufA, nullptr, R, 64, F(55), F(58));
  biconvP(R, 64, F(50), F(53), 64, bufB);
  bnrelu_sign(bufB, nullptr, R, 64, F(56), F(59));
  biconvP(R, 64, F(51), F(54), 64, bufA);
  bnrelu_sign(bufA, bufA, R, 64, F(57), F(60));

  tnet(true, nullptr, 64, ix2, bufB, bufC, tmat);
  {
    long long total = (long long)R * 64;
    k_bmm_eye<<<cdiv(total, 256), 256, 0, stream>>>(bufA, tmat, bufB, Npts, 64, total);
  }
  bnrelu_sign(bufB, nullptr, R, 64, F(47), F(48));

  biconvP(R, 64,  F(61), F(63), 128,  bufC);
  bnrelu_sign(bufC, nullptr, R, 128, F(65), F(67));
  biconvP(R, 128, F(62), F(64), 1024, bufA);
  bnrelu(bufA, R, 1024, F(66), F(68));

  k_maxpool<<<B * 1024, 256, 0, stream>>>(bufA, pool, Npts, 1024, OFF);

  biconv(pool, B, 1024, F(69), F(71), 512, q1);
  bnrelu_sign(q1, nullptr, B, 512, F(73), F(75));
  biconvP(B, 512, F(70), F(72), 256, q2);
  bnrelu_sign(q2, nullptr, B, 256, F(74), F(76));
  biconvP(B, 256, F(77), F(78), 40, (float*)d_out);

#undef F
}
